// Attention_50732153700786
// MI455X (gfx1250) — matrix operand order repaired and checked by reading, not compiled, not run on hardware
//
#include <hip/hip_runtime.h>

#define DEVI __device__ __forceinline__

typedef __bf16 bf16_t;
typedef __attribute__((ext_vector_type(16))) __bf16 v16bf;
typedef __attribute__((ext_vector_type(8)))  __bf16 v8bf;
typedef __attribute__((ext_vector_type(8)))  float  v8f;
typedef __attribute__((ext_vector_type(4)))  unsigned v4u;
typedef __attribute__((ext_vector_type(8)))  unsigned v8u;

constexpr int B_ = 4, N_ = 2048, C_ = 512, H_ = 8, D_ = 64, HID_ = 512;
constexpr int QKV_COLS = 3 * HID_;

DEVI int laneid() { return (int)(threadIdx.x & 31u); }

// ---- WMMA fragment loaders (contiguous k = 16*(lane/16) + s, same in A and B) --------------------

// A operand (16x32 bf16) from row-major source, stride ld (elements).
DEVI v16bf load_frag_a(const bf16_t* __restrict__ p, int ld) {
  const int lane = laneid();
  const int m  = lane & 15;
  const int c0 = (lane & 16) ? 16 : 0;
  const bf16_t* q = p + (size_t)m * ld + c0;
  v8bf lo = *(const v8bf*)(q);
  v8bf hi = *(const v8bf*)(q + 8);
  v16bf f;
#pragma unroll
  for (int i = 0; i < 8; ++i) { f[i] = lo[i]; f[i + 8] = hi[i]; }
  return f;
}

// B operand (32x16 bf16) where B^T is row-major with stride ld: p[n*ld + k] = B[k][n].
DEVI v16bf load_frag_b_rm(const bf16_t* __restrict__ p, int ld) {
  const int lane = laneid();
  const int n  = lane & 15;
  const int k0 = (lane & 16) ? 16 : 0;
  const bf16_t* q = p + (size_t)n * ld + k0;
  v8bf lo = *(const v8bf*)(q);
  v8bf hi = *(const v8bf*)(q + 8);
  v16bf f;
#pragma unroll
  for (int i = 0; i < 8; ++i) { f[i] = lo[i]; f[i + 8] = hi[i]; }
  return f;
}

// B operand (32x16 bf16) where B itself is row-major: p[k*ld + n] = B[k][n].
DEVI v16bf load_frag_b_cm(const bf16_t* __restrict__ p, int ld) {
  const int lane = laneid();
  const int n  = lane & 15;
  const int k0 = (lane & 16) ? 16 : 0;
  v16bf f;
#pragma unroll
  for (int i = 0; i < 16; ++i) f[i] = p[(size_t)(k0 + i) * ld + n];
  return f;
}

DEVI v8f wmma_bf16(v16bf a, v16bf b, v8f c) {
  return __builtin_amdgcn_wmma_f32_16x16x32_bf16(false, a, false, b, (short)0, c,
                                                 false, false);
}

DEVI float row_max16(float v) {
#pragma unroll
  for (int off = 1; off < 16; off <<= 1) v = fmaxf(v, __shfl_xor(v, off, 32));
  return v;
}
DEVI float row_sum16(float v) {
#pragma unroll
  for (int off = 1; off < 16; off <<= 1) v += __shfl_xor(v, off, 32);
  return v;
}

// ---- CDNA5 async / tensor data movers ----------------------------------------

DEVI unsigned lds_addr32(const void* p) {        // AS3 offset lives in low 32 bits
  return (unsigned)(uintptr_t)p;
}

DEVI void async_copy_b128(unsigned lds_off, unsigned voff_bytes, const void* sbase) {
  asm volatile("global_load_async_to_lds_b128 %0, %1, %2"
               :
               : "v"(lds_off), "v"(voff_bytes), "s"(sbase)
               : "memory");
}

DEVI void wait_async_le1() { asm volatile("s_wait_asynccnt 0x1" ::: "memory"); }
DEVI void wait_async_0()   { asm volatile("s_wait_asynccnt 0x0" ::: "memory"); }

// TDM: 1-D contiguous tile copy global->LDS. n8 = tile length in 8-byte units.
// Descriptor bitfields per CDNA5 ISA ch.8 (D# group0 128b, group1 256b).
DEVI void tdm_load_1d(unsigned lds_bytes, const void* gptr, unsigned n8) {
  unsigned long long ga = (unsigned long long)(uintptr_t)gptr;
  v4u g0 = {
      1u,                                                   // count=1 (valid)
      lds_bytes,                                            // lds_addr
      (unsigned)ga,                                         // global_addr[31:0]
      ((unsigned)(ga >> 32) & 0x01ffffffu) | (2u << 30)     // addr[56:32] | type=2
  };
  v8u g1 = {
      3u << 16,                    // data_size=3 (8B), workgroup_mask=0
      (n8 & 0xffffu) << 16,        // tensor_dim0[15:0]
      ((n8 >> 16) & 0xffffu)       // tensor_dim0[31:16]
          | (1u << 16),            // tensor_dim1[15:0] = 1
      (n8 & 0xffffu) << 16,        // tensor_dim1[31:16]=0, tile_dim0 = n8
      1u,                          // tile_dim1 = 1, tile_dim2 = 0
      n8,                          // tensor_dim0_stride[31:0]
      0u,                          // stride[47:32]=0, tensor_dim1_stride[15:0]=0
      0u                           // tensor_dim1_stride[47:16]=0
  };
  asm volatile("tensor_load_to_lds %0, %1" : : "s"(g0), "s"(g1) : "memory");
}

// ---- fp32 -> bf16 conversion -------------------------------------------------

__global__ void cvt_f32_to_bf16(const float* __restrict__ s, bf16_t* __restrict__ d,
                                int n) {
  for (int i = blockIdx.x * blockDim.x + threadIdx.x; i < n;
       i += gridDim.x * blockDim.x)
    d[i] = (bf16_t)s[i];
}

// ---- QKV projection GEMM + scale + RoPE epilogue -----------------------------
__global__ __launch_bounds__(256) void qkv_rope_kernel(
    const bf16_t* __restrict__ xb, const bf16_t* __restrict__ wb,
    bf16_t* __restrict__ qo, bf16_t* __restrict__ ko, bf16_t* __restrict__ vo) {
  const int wave = threadIdx.x >> 5;
  const int lane = laneid();
  const int row0 = blockIdx.x * 16;
  const int col0 = (blockIdx.y * 8 + wave) * 64;
  const bf16_t* A = xb + (size_t)row0 * C_;
  const bf16_t* W = wb + (size_t)col0 * C_;

  v8f acc[4] = {v8f{}, v8f{}, v8f{}, v8f{}};
  // two-stage pipeline with in-place stages (no register rotation movs)
  v16bf a0, a1, b0[4], b1[4];
  a0 = load_frag_a(A, C_);
#pragma unroll
  for (int t = 0; t < 4; ++t) b0[t] = load_frag_b_rm(W + (size_t)(t * 16) * C_, C_);

  for (int k = 0; k < C_; k += 64) {
    a1 = load_frag_a(A + k + 32, C_);
#pragma unroll
    for (int t = 0; t < 4; ++t)
      b1[t] = load_frag_b_rm(W + (size_t)(t * 16) * C_ + k + 32, C_);
#pragma unroll
    for (int t = 0; t < 4; ++t) acc[t] = wmma_bf16(a0, b0[t], acc[t]);
    if (k + 64 < C_) {
      a0 = load_frag_a(A + k + 64, C_);
#pragma unroll
      for (int t = 0; t < 4; ++t)
        b0[t] = load_frag_b_rm(W + (size_t)(t * 16) * C_ + k + 64, C_);
    }
#pragma unroll
    for (int t = 0; t < 4; ++t) acc[t] = wmma_bf16(a1, b1[t], acc[t]);
  }

  const int col   = lane & 15;
  const int rbase = (lane & 16) ? 8 : 0;
#pragma unroll
  for (int t = 0; t < 4; ++t) {
    const int gcol    = col0 + t * 16 + col;
    const int section = gcol >> 9;   // 0=q 1=k 2=v (uniform within the tile)
    const int within  = gcol & 511;
    const int h = within >> 6;
    const int d = within & 63;
    const float inv_freq = __powf(10000.0f, -(float)(d & ~1) * (1.0f / 64.0f));
    const float qscale = (section == 0) ? 0.125f : 1.0f;  // 64^-0.5
    bf16_t* dst = (section == 0) ? qo : (section == 1) ? ko : vo;
#pragma unroll
    for (int r = 0; r < 8; ++r) {
      const int grow = row0 + rbase + r;
      const int bb = grow >> 11;
      const int n  = grow & (N_ - 1);
      float v = acc[t][r] * qscale;
      if (section < 2) {                       // RoPE (wave-uniform branch)
        float partner = __shfl_xor(v, 1, 32);
        float sn, cs;
        __sincosf((float)n * inv_freq, &sn, &cs);
        float rot = (d & 1) ? partner : -partner;
        v = v * cs + rot * sn;
      }
      dst[(((size_t)bb * H_ + h) * N_ + n) * D_ + d] = (bf16_t)v;
    }
  }
}

// ---- Flash attention ---------------------------------------------------------
// Block = 8 waves = 8 query tiles (128 rows) of one (b,h).
// K tiles staged by the Tensor Data Mover (wave0 issues one 1-D descriptor,
// TENSORcnt tracked); V tiles staged by per-lane async b128 copies (ASYNCcnt).
// Both double buffered.
__global__ __launch_bounds__(256) void flash_attn_kernel(
    const bf16_t* __restrict__ qg, const bf16_t* __restrict__ kg,
    const bf16_t* __restrict__ vg, const float* __restrict__ bias,
    bf16_t* __restrict__ og) {
  __shared__ __align__(16) bf16_t ldsK[2][32 * 64];   // 2 x 4KB
  __shared__ __align__(16) bf16_t ldsV[2][32 * 64];   // 2 x 4KB
  __shared__ __align__(16) bf16_t pbuf[8][16 * 32];   // per-wave P staging

  const int tid  = threadIdx.x;
  const int wave = tid >> 5;
  const int lane = tid & 31;
  const int bh   = blockIdx.x >> 4;                // uniform: 16 blocks per (b,h)
  const int qt   = (blockIdx.x & 15) * 8 + wave;
  const int b    = bh >> 3;
  const int h    = bh & 7;

  const bf16_t* Q = qg + ((size_t)bh * N_ + (size_t)qt * 16) * D_;
  const bf16_t* K = kg + (size_t)bh * N_ * D_;     // uniform -> SGPR base
  const bf16_t* V = vg + (size_t)bh * N_ * D_;     // uniform -> SGPR base
  const float* Bp = bias + (size_t)h * N_ * N_ + (size_t)(qt * 16) * N_;

  const v16bf qf0 = load_frag_a(Q, D_);
  const v16bf qf1 = load_frag_a(Q + 32, D_);

  v8f o[4] = {v8f{}, v8f{}, v8f{}, v8f{}};
  float mi[8], li[8];
#pragma unroll
  for (int r = 0; r < 8; ++r) { mi[r] = -1e30f; li[r] = 0.0f; }

  bf16_t* pw = &pbuf[wave][0];
  const int col   = lane & 15;
  const int rbase = (lane & 16) ? 8 : 0;

  const unsigned kOff[2] = {lds_addr32(&ldsK[0][0]), lds_addr32(&ldsK[1][0])};
  const unsigned vOff[2] = {lds_addr32(&ldsV[0][0]), lds_addr32(&ldsV[1][0])};

  // prologue: stage tile 0 into buffer 0
  if (wave == 0) tdm_load_1d(kOff[0], (const char*)K, 512);      // 4KB K tile
  async_copy_b128(vOff[0] + tid * 16, (unsigned)tid * 16u, V);   // 4KB V tile

  const int NIT = N_ / 32;
  for (int it = 0; it < NIT; ++it) {
    const int cur = it & 1;
    __syncthreads();                         // all waves done reading buf[cur^1]
    if (it + 1 < NIT) {                      // prefetch next tile into other buffer
      const unsigned gb = (unsigned)((it + 1) * 4096);
      if (wave == 0) tdm_load_1d(kOff[cur ^ 1], (const char*)K + gb, 512);
      async_copy_b128(vOff[cur ^ 1] + tid * 16, gb + (unsigned)tid * 16u, V);
      __builtin_amdgcn_s_wait_tensorcnt((short)1);   // oldest (current K) done
      wait_async_le1();                              // oldest (current V) done
    } else {
      __builtin_amdgcn_s_wait_tensorcnt((short)0);
      wait_async_0();
    }
    __syncthreads();                         // current tile visible to all waves

    const bf16_t* Kt = &ldsK[cur][0];
    const bf16_t* Vt = &ldsV[cur][0];
    const int kt = it * 32;

    // S tiles (16q x 16k each): Q @ K^T, K fragments from LDS (ds_load_b128)
    v8f s0 = {}, s1 = {};
    {
      v16bf k0a = load_frag_b_rm(Kt, D_);
      v16bf k0b = load_frag_b_rm(Kt + 32, D_);
      s0 = wmma_bf16(qf0, k0a, s0);
      s0 = wmma_bf16(qf1, k0b, s0);
      v16bf k1a = load_frag_b_rm(Kt + 16 * D_, D_);
      v16bf k1b = load_frag_b_rm(Kt + 16 * D_ + 32, D_);
      s1 = wmma_bf16(qf0, k1a, s1);
      s1 = wmma_bf16(qf1, k1b, s1);
    }
    // bias add + online softmax (f32); P (bf16) written to per-wave LDS tile
#pragma unroll
    for (int r = 0; r < 8; ++r) {
      const float* br = Bp + (size_t)(rbase + r) * N_ + kt;
      float sv0 = s0[r] + br[col];
      float sv1 = s1[r] + br[16 + col];
      float mloc = row_max16(fmaxf(sv0, sv1));
      float mnew = fmaxf(mi[r], mloc);
      float corr = __expf(mi[r] - mnew);
      float e0 = __expf(sv0 - mnew);
      float e1 = __expf(sv1 - mnew);
      li[r] = li[r] * corr + row_sum16(e0 + e1);
      mi[r] = mnew;
#pragma unroll
      for (int t = 0; t < 4; ++t) o[t][r] *= corr;
      pw[(rbase + r) * 32 + col]      = (bf16_t)e0;
      pw[(rbase + r) * 32 + 16 + col] = (bf16_t)e1;
    }
    asm volatile("s_wait_dscnt 0x0" ::: "memory");
    __builtin_amdgcn_wave_barrier();
    v16bf pf = load_frag_a(pw, 32);          // D-layout -> A-layout via LDS
    // P(16x32) @ V(32x64): V^T fragments gathered from LDS
#pragma unroll
    for (int t = 0; t < 4; ++t) {
      v16bf vf = load_frag_b_cm(Vt + t * 16, D_);
      o[t] = wmma_bf16(pf, vf, o[t]);
    }
    __builtin_amdgcn_wave_barrier();
  }

#pragma unroll
  for (int r = 0; r < 8; ++r) {
    const float inv = 1.0f / li[r];
    const int n = qt * 16 + rbase + r;
#pragma unroll
    for (int t = 0; t < 4; ++t)
      og[((size_t)b * N_ + n) * HID_ + h * D_ + t * 16 + col] =
          (bf16_t)(o[t][r] * inv);
  }
}

// ---- Output projection GEMM: out = attn(8192x512) @ w_out^T(512x512), f32 out
__global__ __launch_bounds__(256) void out_gemm_kernel(
    const bf16_t* __restrict__ ab, const bf16_t* __restrict__ wb,
    float* __restrict__ out) {
  const int wave = threadIdx.x >> 5;
  const int lane = laneid();
  const int row0 = blockIdx.x * 16;
  const int col0 = wave * 64;
  const bf16_t* A = ab + (size_t)row0 * HID_;
  const bf16_t* W = wb + (size_t)col0 * HID_;
  v8f acc[4] = {v8f{}, v8f{}, v8f{}, v8f{}};

  v16bf a0, a1, b0[4], b1[4];
  a0 = load_frag_a(A, HID_);
#pragma unroll
  for (int t = 0; t < 4; ++t) b0[t] = load_frag_b_rm(W + (size_t)(t * 16) * HID_, HID_);

  for (int k = 0; k < HID_; k += 64) {
    a1 = load_frag_a(A + k + 32, HID_);
#pragma unroll
    for (int t = 0; t < 4; ++t)
      b1[t] = load_frag_b_rm(W + (size_t)(t * 16) * HID_ + k + 32, HID_);
#pragma unroll
    for (int t = 0; t < 4; ++t) acc[t] = wmma_bf16(a0, b0[t], acc[t]);
    if (k + 64 < HID_) {
      a0 = load_frag_a(A + k + 64, HID_);
#pragma unroll
      for (int t = 0; t < 4; ++t)
        b0[t] = load_frag_b_rm(W + (size_t)(t * 16) * HID_ + k + 64, HID_);
    }
#pragma unroll
    for (int t = 0; t < 4; ++t) acc[t] = wmma_bf16(a1, b1[t], acc[t]);
  }

  const int col   = lane & 15;
  const int rbase = (lane & 16) ? 8 : 0;
#pragma unroll
  for (int t = 0; t < 4; ++t)
#pragma unroll
    for (int r = 0; r < 8; ++r)
      out[(size_t)(row0 + rbase + r) * HID_ + col0 + t * 16 + col] = acc[t][r];
}

// ---- host launcher -----------------------------------------------------------

extern "C" void kernel_launch(void* const* d_in, const int* in_sizes, int n_in,
                              void* d_out, int out_size, void* d_ws, size_t ws_size,
                              hipStream_t stream) {
  (void)in_sizes; (void)n_in; (void)out_size; (void)ws_size;
  const float* x   = (const float*)d_in[0];
  const float* pb  = (const float*)d_in[1];
  const float* wqk = (const float*)d_in[2];
  const float* wo  = (const float*)d_in[3];
  float* out = (float*)d_out;

  char* ws = (char*)d_ws;
  size_t off = 0;
  auto alloc = [&](size_t bytes) {
    void* p = ws + off;
    off += (bytes + 255) & ~(size_t)255;
    return p;
  };
  bf16_t* xb  = (bf16_t*)alloc((size_t)B_ * N_ * C_ * 2);
  bf16_t* wqb = (bf16_t*)alloc((size_t)QKV_COLS * C_ * 2);
  bf16_t* wob = (bf16_t*)alloc((size_t)HID_ * HID_ * 2);
  bf16_t* q   = (bf16_t*)alloc((size_t)B_ * H_ * N_ * D_ * 2);
  bf16_t* k   = (bf16_t*)alloc((size_t)B_ * H_ * N_ * D_ * 2);
  bf16_t* v   = (bf16_t*)alloc((size_t)B_ * H_ * N_ * D_ * 2);
  bf16_t* ao  = (bf16_t*)alloc((size_t)B_ * N_ * HID_ * 2);

  cvt_f32_to_bf16<<<4096, 256, 0, stream>>>(x, xb, B_ * N_ * C_);
  cvt_f32_to_bf16<<<1024, 256, 0, stream>>>(wqk, wqb, QKV_COLS * C_);
  cvt_f32_to_bf16<<<512, 256, 0, stream>>>(wo, wob, HID_ * HID_);

  qkv_rope_kernel<<<dim3((B_ * N_) / 16, 3), 256, 0, stream>>>(xb, wqb, q, k, v);
  flash_attn_kernel<<<(B_ * H_ * (N_ / 16)) / 8, 256, 0, stream>>>(q, k, v, pb, ao);
  out_gemm_kernel<<<(B_ * N_) / 16, 256, 0, stream>>>(ao, wob, out);
}
